// VisionTransformerLocal_46308337385574
// MI455X (gfx1250) — matrix operand order repaired and checked by reading, not compiled, not run on hardware
//
#include <hip/hip_runtime.h>
#include <hip/hip_bf16.h>
#include <math.h>

// ---------------------------------------------------------------------------
// MI455X (gfx1250) ViT forward.  wave32, WMMA f32_16x16x32_f16 everywhere.
// ---------------------------------------------------------------------------

typedef __attribute__((ext_vector_type(16))) _Float16 v16h;
typedef __attribute__((ext_vector_type(8)))  float    v8f;

union F16Frag { v16h v; unsigned int u[8]; _Float16 h[16]; };

#define WMMA(a, b, c) __builtin_amdgcn_wmma_f32_16x16x32_f16( \
    false, (a), false, (b), (short)0, (c), false, false)

// problem dims
static constexpr int D_B  = 4;
static constexpr int D_T  = 784;     // 28*28 patches
static constexpr int D_G  = 28;
static constexpr int D_E  = 768;
static constexpr int D_HD = 3072;
static constexpr int D_NH = 12;
static constexpr int D_DH = 64;
static constexpr int D_M  = D_B * D_T;   // 3136 rows of activations

__device__ __forceinline__ int iabs_(int v) { return v < 0 ? -v : v; }

// ---------------------------------------------------------------------------
// Generic GEMM:  C[M,N] = A[M,K] (f32) @ W[N,K]^T (f32, torch layout) + bias
// EPI: 0 = bias (f32 out), 1 = bias + residual R, 2 = bias + exact GELU,
//      3 = bias, output written as f16 (for the QKV buffer).
// Block: 256 thr = 8 waves; tile 128x64; wave tile 32x32 (2x2 WMMA frags).
// ---------------------------------------------------------------------------
static constexpr int BM = 128, BN = 64, BK = 32;
static constexpr int LDA_S = 34;   // padded f16 row stride for A tile
static constexpr int LDB_S = 66;   // padded f16 row stride for B^T tile

template <int EPI>
__global__ __launch_bounds__(256)
void gemm_wmma(const float* __restrict__ A, const float* __restrict__ W,
               const float* __restrict__ bias, const float* R,
               float* C, int M, int N, int K)
{
    __shared__ __attribute__((aligned(16))) _Float16 sA[BM * LDA_S];
    __shared__ __attribute__((aligned(16))) _Float16 sB[BK * LDB_S];

    const int tid  = threadIdx.x;
    const int lane = tid & 31;
    const int wid  = tid >> 5;
    const int wm   = wid & 3;           // wave row-slab (4 x 32 rows)
    const int wn   = wid >> 2;          // wave col-slab (2 x 32 cols)
    const int m0   = blockIdx.x * BM;
    const int n0   = blockIdx.y * BN;
    const int l15  = lane & 15;
    const int half = lane >> 4;

    v8f acc[2][2];
#pragma unroll
    for (int a = 0; a < 2; ++a)
#pragma unroll
        for (int b = 0; b < 2; ++b)
#pragma unroll
            for (int j = 0; j < 8; ++j) acc[a][b][j] = 0.0f;

    for (int k0 = 0; k0 < K; k0 += BK) {
        // --- stage A tile (f32 -> f16), zero-padded past M ---
        {
            const int r  = tid >> 1;
            const int cg = (tid & 1) * 16;
            const int gm = m0 + r;
            const bool ok = gm < M;
            const float* ap = A + (size_t)gm * K + k0 + cg;
#pragma unroll
            for (int i = 0; i < 16; ++i)
                sA[r * LDA_S + cg + i] = (_Float16)(ok ? ap[i] : 0.0f);
        }
        // --- stage W tile transposed: sB[k][n] = W[n0+n][k0+k] ---
        {
            const int n  = tid & 63;
            const int kg = (tid >> 6) * 8;
            const float* wp = W + (size_t)(n0 + n) * K + k0 + kg;
#pragma unroll
            for (int i = 0; i < 8; ++i)
                sB[(kg + i) * LDB_S + n] = (_Float16)wp[i];
        }
        __syncthreads();

        F16Frag af[2], bf[2];
#pragma unroll
        for (int fm = 0; fm < 2; ++fm) {
            const int row = wm * 32 + fm * 16 + l15;
#pragma unroll
            for (int j = 0; j < 8; ++j) {
                const int kk = 2 * j + 8 * half + ((j >= 4) ? 8 : 0);
                af[fm].u[j] = *(const unsigned int*)&sA[row * LDA_S + kk];
            }
        }
#pragma unroll
        for (int fn = 0; fn < 2; ++fn) {
            const int nc = wn * 32 + fn * 16;
#pragma unroll
            for (int i = 0; i < 16; ++i)
                bf[fn].h[i] = sB[((i < 8 ? i : i + 8) + 8 * half) * LDB_S + nc + l15];
        }
#pragma unroll
        for (int fm = 0; fm < 2; ++fm)
#pragma unroll
            for (int fn = 0; fn < 2; ++fn)
                acc[fm][fn] = WMMA(af[fm].v, bf[fn].v, acc[fm][fn]);
        __syncthreads();
    }

    // --- epilogue ---
#pragma unroll
    for (int fm = 0; fm < 2; ++fm)
#pragma unroll
        for (int fn = 0; fn < 2; ++fn)
#pragma unroll
            for (int j = 0; j < 8; ++j) {
                const int gm = m0 + wm * 32 + fm * 16 + j + 8 * half;
                const int gn = n0 + wn * 32 + fn * 16 + l15;
                if (gm < M) {
                    float v = acc[fm][fn][j] + bias[gn];
                    if (EPI == 1) v += R[(size_t)gm * N + gn];
                    if (EPI == 2) v = 0.5f * v * (1.0f + erff(v * 0.70710678118654752f));
                    if (EPI == 3)
                        ((_Float16*)C)[(size_t)gm * N + gn] = (_Float16)v;
                    else
                        C[(size_t)gm * N + gn] = v;
                }
            }
}

// ---------------------------------------------------------------------------
// Flash attention with the (inverted) local-window mask.
// qkv is f16 [B*T, 3E] (q | k | v); o is f32 [B*T, E].
// One wave per (batch, head, 16-row q block); key blocks of 32.
// Per block: lane-per-key b128 loads stage K^T and V in per-wave LDS tiles,
// then 4 S-WMMAs + 4 O-WMMAs; P bounced via LDS (D-layout -> A-layout).
// ---------------------------------------------------------------------------
static constexpr float NEG_BIG = -1e30f;
static constexpr int KT_S = 34;   // sKT row stride (f16): [dh 0..63][key 0..31]
static constexpr int V_S  = 66;   // sV  row stride (f16): [key 0..31][dh 0..63]

__global__ __launch_bounds__(256)
void attn_kernel(const _Float16* __restrict__ qkv, float* __restrict__ o)
{
    const int lane = threadIdx.x & 31;
    const int wid  = threadIdx.x >> 5;
    const int task = blockIdx.x * 8 + wid;          // 49*12*4 = 2352 tasks
    const int qb   = task % 49;
    const int hh   = (task / 49) % D_NH;
    const int bb   = task / (49 * D_NH);

    const int l15  = lane & 15;
    const int half = lane >> 4;
    const size_t rs = 3 * (size_t)D_E;              // 2304 halves per token

    __shared__ __attribute__((aligned(16))) _Float16 sKT[8][64 * KT_S];
    __shared__ __attribute__((aligned(16))) _Float16 sV [8][32 * V_S];
    __shared__ __attribute__((aligned(16))) _Float16 sP [8][16 * 34];

    // Q fragments (A-layout) as raw packed-half u32 loads; the 1/sqrt(DH)
    // scale is folded into S after the WMMA.
    F16Frag qf[2];
    const _Float16* qrow =
        qkv + ((size_t)bb * D_T + qb * 16 + l15) * rs + hh * D_DH;
#pragma unroll
    for (int s = 0; s < 2; ++s)
#pragma unroll
        for (int j = 0; j < 8; ++j) {
            const int kk = 2 * j + 8 * half + ((j >= 4) ? 8 : 0) + 32 * s;
            qf[s].u[j] = *(const unsigned int*)(qrow + kk);
        }

    float mrun[8], lrun[8];
    v8f ofr[4];
#pragma unroll
    for (int j = 0; j < 8; ++j) { mrun[j] = NEG_BIG; lrun[j] = 0.0f; }
#pragma unroll
    for (int f = 0; f < 4; ++f)
#pragma unroll
        for (int j = 0; j < 8; ++j) ofr[f][j] = 0.0f;

    for (int kb = 0; kb < 25; ++kb) {
        const int kB = kb * 32;

        // ---- stage K^T (transposed) and V for this 32-key block ----
        {
            const int key  = kB + lane;
            const int keyc = (key < D_T) ? key : 0;   // clamp; masked later
            const _Float16* kr =
                qkv + ((size_t)bb * D_T + keyc) * rs + D_E + hh * D_DH;
            const uint4* k4 = (const uint4*)kr;
            const uint4* v4 = (const uint4*)(kr + D_E);
#pragma unroll
            for (int i = 0; i < 8; ++i) {
                uint4 kq = k4[i];
                uint4 vq = v4[i];
                const _Float16* kh = (const _Float16*)&kq;
#pragma unroll
                for (int e = 0; e < 8; ++e)
                    sKT[wid][(i * 8 + e) * KT_S + lane] = kh[e];
                unsigned int* vdst = (unsigned int*)&sV[wid][lane * V_S + i * 8];
                const unsigned int* vsrc = (const unsigned int*)&vq;
#pragma unroll
                for (int e = 0; e < 4; ++e) vdst[e] = vsrc[e];
            }
        }
        __syncthreads();

        // ---- S = Q @ K^T (scale applied after), two 16-key columns ----
        v8f sf[2];
#pragma unroll
        for (int f = 0; f < 2; ++f) {
            v8f z;
#pragma unroll
            for (int j = 0; j < 8; ++j) z[j] = 0.0f;
#pragma unroll
            for (int s = 0; s < 2; ++s) {
                F16Frag kf;                      // B-frag: lane = key col, elements = dh
#pragma unroll
                for (int i = 0; i < 16; ++i)
                    kf.h[i] =
                        sKT[wid][(32 * s + (i < 8 ? i : i + 8) + 8 * half) * KT_S + f * 16 + l15];
                z = WMMA(qf[s].v, kf.v, z);
            }
            sf[f] = z;
        }

        // ---- mask + online softmax (rows on VGPR index j, half) ----
        float p0[8], p1[8], alpha[8];
        const int k0g = kB + l15;
        const int k1g = kB + 16 + l15;
        const int r0 = k0g / D_G, c0 = k0g % D_G;
        const int r1 = k1g / D_G, c1 = k1g % D_G;
#pragma unroll
        for (int j = 0; j < 8; ++j) {
            const int qg = qb * 16 + j + 8 * half;
            const int rq = qg / D_G, cq = qg % D_G;
            float s0 = sf[0][j] * 0.125f, s1 = sf[1][j] * 0.125f;
            const bool bad0 = (k0g >= D_T) || (iabs_(rq - r0) <= 4 && iabs_(cq - c0) <= 4);
            const bool bad1 = (k1g >= D_T) || (iabs_(rq - r1) <= 4 && iabs_(cq - c1) <= 4);
            s0 = bad0 ? NEG_BIG : s0;
            s1 = bad1 ? NEG_BIG : s1;
            float t = fmaxf(s0, s1);
            t = fmaxf(t, __shfl_xor(t, 1));
            t = fmaxf(t, __shfl_xor(t, 2));
            t = fmaxf(t, __shfl_xor(t, 4));
            t = fmaxf(t, __shfl_xor(t, 8));
            const float mn  = fmaxf(mrun[j], t);
            const bool  any = mn > -1e29f;
            const float a   = any ? __expf(mrun[j] - mn) : 1.0f;
            const float e0  = any ? __expf(s0 - mn) : 0.0f;
            const float e1  = any ? __expf(s1 - mn) : 0.0f;
            float rsum = e0 + e1;
            rsum += __shfl_xor(rsum, 1);
            rsum += __shfl_xor(rsum, 2);
            rsum += __shfl_xor(rsum, 4);
            rsum += __shfl_xor(rsum, 8);
            lrun[j] = lrun[j] * a + rsum;
            mrun[j] = mn;
            alpha[j] = a;
            p0[j] = e0;
            p1[j] = e1;
        }
#pragma unroll
        for (int f = 0; f < 4; ++f)
#pragma unroll
            for (int j = 0; j < 8; ++j) ofr[f][j] *= alpha[j];

        // ---- bounce P through LDS: D-layout -> A-layout ----
#pragma unroll
        for (int j = 0; j < 8; ++j) {
            sP[wid][(j + 8 * half) * 34 + l15]      = (_Float16)p0[j];
            sP[wid][(j + 8 * half) * 34 + 16 + l15] = (_Float16)p1[j];
        }
        __syncthreads();
        F16Frag pf;
#pragma unroll
        for (int j = 0; j < 8; ++j) {
            const int kk = 2 * j + 8 * half + ((j >= 4) ? 8 : 0);
            pf.u[j] = *(const unsigned int*)&sP[wid][l15 * 34 + kk];
        }

        // ---- O += P @ V (V B-frag: lane = dh col, elements = key) ----
#pragma unroll
        for (int f = 0; f < 4; ++f) {
            F16Frag vf;
#pragma unroll
            for (int i = 0; i < 16; ++i)
                vf.h[i] =
                    sV[wid][((i < 8 ? i : i + 8) + 8 * half) * V_S + f * 16 + l15];
            ofr[f] = WMMA(pf.v, vf.v, ofr[f]);
        }
    }

    // ---- normalize + store ----
#pragma unroll
    for (int j = 0; j < 8; ++j) {
        const float inv = (lrun[j] > 0.0f) ? (1.0f / lrun[j]) : 0.0f;
        const int row = qb * 16 + j + 8 * half;
        const size_t base = ((size_t)bb * D_T + row) * D_E + hh * D_DH;
#pragma unroll
        for (int f = 0; f < 4; ++f)
            o[base + f * 16 + l15] = ofr[f][j] * inv;
    }
}

// ---------------------------------------------------------------------------
// Elementwise / reduction kernels
// ---------------------------------------------------------------------------
__device__ __forceinline__ float block_sum(float v, float* red)
{
    const int tid = threadIdx.x;
    red[tid] = v;
    __syncthreads();
    for (int off = 128; off > 0; off >>= 1) {
        if (tid < off) red[tid] += red[tid + off];
        __syncthreads();
    }
    const float r = red[0];
    __syncthreads();
    return r;
}

__global__ __launch_bounds__(256)
void ln_kernel(const float* __restrict__ x, const float* __restrict__ g,
               const float* __restrict__ b, float* __restrict__ y)
{
    __shared__ float red[256];
    const int row = blockIdx.x;
    const int tid = threadIdx.x;
    const float* xr = x + (size_t)row * D_E;
    const float v0 = xr[tid], v1 = xr[tid + 256], v2 = xr[tid + 512];
    const float mean = block_sum(v0 + v1 + v2, red) * (1.0f / D_E);
    const float d0 = v0 - mean, d1 = v1 - mean, d2 = v2 - mean;
    const float var = block_sum(d0 * d0 + d1 * d1 + d2 * d2, red) * (1.0f / D_E);
    const float inv = rsqrtf(var + 1e-5f);
    float* yr = y + (size_t)row * D_E;
    yr[tid]       = d0 * inv * g[tid]       + b[tid];
    yr[tid + 256] = d1 * inv * g[tid + 256] + b[tid + 256];
    yr[tid + 512] = d2 * inv * g[tid + 512] + b[tid + 512];
}

__global__ __launch_bounds__(256)
void im2col(const float* __restrict__ x, float* __restrict__ patches)
{
    const size_t idx = (size_t)blockIdx.x * 256 + threadIdx.x;
    if (idx >= (size_t)D_M * D_E) return;
    const int k = (int)(idx % D_E);
    const int m = (int)(idx / D_E);
    const int b = m / D_T, t = m % D_T;
    const int r = t / D_G, cc = t % D_G;
    const int c = k / 256, rem = k % 256;
    const int py = rem / 16, px = rem % 16;
    patches[idx] = x[(((size_t)b * 448 + r * 16 + py) * 448 + cc * 16 + px) * 3 + c];
}

__global__ __launch_bounds__(256)
void add_pos(float* __restrict__ h, const float* __restrict__ pos)
{
    const size_t idx = (size_t)blockIdx.x * 256 + threadIdx.x;
    if (idx >= (size_t)D_M * D_E) return;
    const int e = (int)(idx % D_E);
    const int t = (int)((idx / D_E) % D_T);
    h[idx] += pos[(size_t)t * D_E + e];
}

__global__ __launch_bounds__(256)
void head_kernel(const float* __restrict__ h, const float* __restrict__ gw,
                 const float* __restrict__ gb, const float* __restrict__ hw,
                 const float* __restrict__ hb, const float* __restrict__ sm,
                 const float* __restrict__ ss, float* __restrict__ out)
{
    __shared__ float red[256];
    const int row = blockIdx.x;
    const int tid = threadIdx.x;
    const float* xr = h + (size_t)row * D_E;
    const float v0 = xr[tid], v1 = xr[tid + 256], v2 = xr[tid + 512];
    const float mean = block_sum(v0 + v1 + v2, red) * (1.0f / D_E);
    const float d0 = v0 - mean, d1 = v1 - mean, d2 = v2 - mean;
    const float var = block_sum(d0 * d0 + d1 * d1 + d2 * d2, red) * (1.0f / D_E);
    const float inv = rsqrtf(var + 1e-5f);
    float part = (d0 * inv * gw[tid]       + gb[tid])       * hw[tid]
               + (d1 * inv * gw[tid + 256] + gb[tid + 256]) * hw[tid + 256]
               + (d2 * inv * gw[tid + 512] + gb[tid + 512]) * hw[tid + 512];
    const float dot = block_sum(part, red);
    if (tid == 0) {
        const float logit = dot + hb[0];
        const float z = logit * ss[0] + sm[0];
        float flux = __expf(z * 2.302585092994046f) - 1e-8f;   // 10^z
        flux = fminf(fmaxf(flux, 1e-15f), 1.0f);
        out[4 + row] = flux;
    }
}

__global__ __launch_bounds__(256)
void flux_reduce(float* out)
{
    __shared__ float red[256];
    const int b = blockIdx.x;
    float s = 0.0f;
    for (int i = threadIdx.x; i < D_T; i += 256) s += out[4 + b * D_T + i];
    const float tot = block_sum(s, red);
    if (threadIdx.x == 0) out[b] = fmaxf(tot, 1e-15f);
}

// ---------------------------------------------------------------------------
// Driver
// ---------------------------------------------------------------------------
extern "C" void kernel_launch(void* const* d_in, const int* in_sizes, int n_in,
                              void* d_out, int out_size, void* d_ws, size_t ws_size,
                              hipStream_t stream)
{
    const float* x         = (const float*)d_in[0];
    const float* sxr_mean  = (const float*)d_in[1];
    const float* sxr_std   = (const float*)d_in[2];
    const float* input_w   = (const float*)d_in[3];
    const float* input_b   = (const float*)d_in[4];
    const float* pos_emb   = (const float*)d_in[5];
    const float* ln1_w     = (const float*)d_in[6];
    const float* ln1_b     = (const float*)d_in[7];
    const float* in_w      = (const float*)d_in[8];
    const float* in_b      = (const float*)d_in[9];
    const float* out_w     = (const float*)d_in[10];
    const float* out_b     = (const float*)d_in[11];
    const float* ln2_w     = (const float*)d_in[12];
    const float* ln2_b     = (const float*)d_in[13];
    const float* w1        = (const float*)d_in[14];
    const float* b1        = (const float*)d_in[15];
    const float* w2        = (const float*)d_in[16];
    const float* b2        = (const float*)d_in[17];
    const float* head_ln_w = (const float*)d_in[18];
    const float* head_ln_b = (const float*)d_in[19];
    const float* head_w    = (const float*)d_in[20];
    const float* head_b    = (const float*)d_in[21];
    float* out = (float*)d_out;

    float* ws = (float*)d_ws;
    const size_t SZ_H = (size_t)D_M * D_E;                 // 2,408,448
    float* buf_h   = ws;
    float* buf_y   = ws + SZ_H;
    float* buf_o   = ws + 2 * SZ_H;
    float* buf_qkv = ws + 3 * SZ_H;                        // f16 [D_M, 3E]
    float* buf_mid = ws + 3 * SZ_H + (size_t)D_M * 3 * D_E;// D_M * HD (also im2col)

    const int elems = D_M * D_E;
    const int eblk  = (elems + 255) / 256;

    im2col<<<eblk, 256, 0, stream>>>(x, buf_mid);
    gemm_wmma<0><<<dim3(25, 12), 256, 0, stream>>>(buf_mid, input_w, input_b,
                                                   nullptr, buf_h, D_M, D_E, D_E);
    add_pos<<<eblk, 256, 0, stream>>>(buf_h, pos_emb);

    for (int l = 0; l < 6; ++l) {
        ln_kernel<<<D_M, 256, 0, stream>>>(buf_h, ln1_w + l * D_E, ln1_b + l * D_E, buf_y);
        gemm_wmma<3><<<dim3(25, 36), 256, 0, stream>>>(
            buf_y, in_w + (size_t)l * 3 * D_E * D_E, in_b + l * 3 * D_E,
            nullptr, buf_qkv, D_M, 3 * D_E, D_E);          // f16 qkv out
        attn_kernel<<<294, 256, 0, stream>>>((const _Float16*)buf_qkv, buf_o);
        gemm_wmma<1><<<dim3(25, 12), 256, 0, stream>>>(
            buf_o, out_w + (size_t)l * D_E * D_E, out_b + l * D_E,
            buf_h, buf_h, D_M, D_E, D_E);
        ln_kernel<<<D_M, 256, 0, stream>>>(buf_h, ln2_w + l * D_E, ln2_b + l * D_E, buf_y);
        gemm_wmma<2><<<dim3(25, 48), 256, 0, stream>>>(
            buf_y, w1 + (size_t)l * D_HD * D_E, b1 + l * D_HD,
            nullptr, buf_mid, D_M, D_HD, D_E);
        gemm_wmma<1><<<dim3(25, 12), 256, 0, stream>>>(
            buf_mid, w2 + (size_t)l * D_E * D_HD, b2 + l * D_E,
            buf_h, buf_h, D_M, D_E, D_HD);
    }

    head_kernel<<<D_M, 256, 0, stream>>>(buf_h, head_ln_w, head_ln_b,
                                         head_w, head_b, sxr_mean, sxr_std, out);
    flux_reduce<<<D_B, 256, 0, stream>>>(out);
}
